// PDELoss_26310969655698
// MI455X (gfx1250) — matrix operand order repaired and checked by reading, not compiled, not run on hardware
//
#include <hip/hip_runtime.h>
#include <hip/hip_bf16.h>

// Problem constants from the reference
#define NN 8192
#define DX 64
#define DZ 128
#define DY 16
#define HH 512

#define M_TILE 16          // rows of N per workgroup
#define NWAVES 8           // 256 threads / wave32

typedef __attribute__((ext_vector_type(2)))  float  v2f;
typedef __attribute__((ext_vector_type(8)))  float  v8f;
typedef __attribute__((ext_vector_type(8)))  __bf16 v8bf;
typedef __attribute__((ext_vector_type(16))) __bf16 v16bf;

// CDNA5 hardware V_TANH_F32 (TRANS pipe, co-executes with WMMA).
#if __has_builtin(__builtin_amdgcn_tanhf)
static __device__ __forceinline__ float fast_tanh(float x) { return __builtin_amdgcn_tanhf(x); }
#elif __has_builtin(__builtin_amdgcn_tanh_f32)
static __device__ __forceinline__ float fast_tanh(float x) { return __builtin_amdgcn_tanh_f32(x); }
#else
static __device__ __forceinline__ float fast_tanh(float x) { return tanhf(x); }
#endif

static __device__ __forceinline__ v8f wmma_f32_k4(v2f a, v2f b, v8f c) {
    return __builtin_amdgcn_wmma_f32_16x16x4_f32(
        false, a, false, b, (short)0, c, false, false);
}
static __device__ __forceinline__ v8f wmma_bf16_k32(v16bf a, v16bf b, v8f c) {
    return __builtin_amdgcn_wmma_f32_16x16x32_bf16(
        false, a, false, b, (short)0, c, false, false);
}
static __device__ __forceinline__ v16bf cat8(v8bf a, v8bf b) {
    return __builtin_shufflevector(a, b, 0,1,2,3,4,5,6,7,8,9,10,11,12,13,14,15);
}
static __device__ __forceinline__ void splitf(float x, __bf16& h, __bf16& l) {
    h = (__bf16)x;
    l = (__bf16)(x - (float)h);
}

// ---------------- weight packing (f32 -> bf16 hi/lo K-panels of 32) ----------------
// Panel layout: dst[(p*Ncols + n)*32 + (k & 31)], p = k >> 5.
// Lane n's B-fragment for panel p = 8 bf16 at ((p*Ncols+n)*32) + (half?8:0), plus 8 more at +16.
__global__ void pack_panels(const float* __restrict__ W, __bf16* __restrict__ hi,
                            __bf16* __restrict__ lo, int K, int Ncols) {
    int idx = blockIdx.x * blockDim.x + threadIdx.x;
    if (idx >= K * Ncols) return;
    int k = idx / Ncols, n = idx % Ncols;
    float w = W[idx];
    __bf16 h, l; splitf(w, h, l);
    int dst = (((k >> 5) * Ncols + n) << 5) + (k & 31);
    hi[dst] = h;
    lo[dst] = l;
}

// Padded LDS strides
#define XSB_STRIDE 72      // bf16 X planes: 16 x 64
#define GSB_STRIDE 528     // bf16 G planes: 16 x 512
#define ZS_STRIDE 132      // f32 z tile: 16 x 128
#define YS_STRIDE 20       // f32 y tile: 16 x 16

// =======================================================================
// bf16x3 fused kernel (primary path)
// =======================================================================
__global__ __launch_bounds__(256) void pde_loss_fused_bf16x3(
    const float*  __restrict__ z_pred,   // (N, DZ)
    const float*  __restrict__ x_label,  // (N, DX)
    const float*  __restrict__ y_label,  // (N, DY)
    const float*  __restrict__ b1,       // (H,)
    const __bf16* __restrict__ W1hi, const __bf16* __restrict__ W1lo, // packed (2,HH,32)
    const __bf16* __restrict__ W2hi, const __bf16* __restrict__ W2lo, // packed (16,DZ,32)
    const float*  __restrict__ Amat,     // (DZ, DZ)
    const float*  __restrict__ Bmat,     // (DZ, DY)
    float* __restrict__ out)
{
    __shared__ __bf16 Xhi [M_TILE * XSB_STRIDE];
    __shared__ __bf16 Xlo [M_TILE * XSB_STRIDE];
    __shared__ __bf16 Xdhi[M_TILE * XSB_STRIDE];
    __shared__ __bf16 Xdlo[M_TILE * XSB_STRIDE];
    __shared__ __bf16 Ghi [M_TILE * GSB_STRIDE];
    __shared__ __bf16 Glo [M_TILE * GSB_STRIDE];
    __shared__ float  Zs  [M_TILE * ZS_STRIDE];
    __shared__ float  Ys  [M_TILE * YS_STRIDE];
    __shared__ float  rowsum[M_TILE];

    const int tid  = threadIdx.x;
    const int wave = tid >> 5;
    const int lane = tid & 31;
    const int m    = lane & 15;
    const bool hiHalf = (lane >= 16);
    const int row0 = blockIdx.x * M_TILE;

    // ---------------- Stage 0: stage tiles into LDS ----------------
    #pragma unroll
    for (int i = 0; i < 4; ++i) {               // X tile: 16x64
        int idx = tid + i * 256;
        int r = idx >> 6, c = idx & 63;
        float x  = x_label[(row0 + r) * DX + c];
        float xd = fast_tanh(x);
        __bf16 h, l;
        splitf(x,  h, l); Xhi [r * XSB_STRIDE + c] = h; Xlo [r * XSB_STRIDE + c] = l;
        splitf(xd, h, l); Xdhi[r * XSB_STRIDE + c] = h; Xdlo[r * XSB_STRIDE + c] = l;
    }
    #pragma unroll
    for (int i = 0; i < 8; ++i) {               // z tile: 16x128
        int idx = tid + i * 256;
        int r = idx >> 7, c = idx & 127;
        Zs[r * ZS_STRIDE + c] = z_pred[(row0 + r) * DZ + c];
    }
    {                                           // y tile: 16x16
        int r = tid >> 4, c = tid & 15;
        Ys[r * YS_STRIDE + c] = y_label[(row0 + r) * DY + c];
    }
    if (tid < M_TILE) rowsum[tid] = 0.0f;
    __syncthreads();

    // A-fragment chunk offsets (16-bit A layout, ISA 7.12.2):
    // lanes 0-15: K = {c..c+7, c+16..c+23}; lanes 16-31: K = {c+8..c+15, c+24..c+31}
    const int aoff = hiHalf ? 8 : 0;
    const int boff = hiHalf ? 8 : 0;    // B fragment: same K chunks as A (boff..+7, boff+16..+23)

    // ---------------- Stage A: G = tanh'(X*W1+b1) .* (tanh(X)*W1) ----------------
    // X fragments are invariant across the 4 H-column tiles: load them ONCE.
    v16bf axh[DX / 32], axl[DX / 32], avh[DX / 32], avl[DX / 32];
    #pragma unroll
    for (int p = 0; p < DX / 32; ++p) {
        const int kb = p * 32 + aoff;
        axh[p] = cat8(*(const v8bf*)&Xhi [m*XSB_STRIDE + kb],
                      *(const v8bf*)&Xhi [m*XSB_STRIDE + kb + 16]);
        axl[p] = cat8(*(const v8bf*)&Xlo [m*XSB_STRIDE + kb],
                      *(const v8bf*)&Xlo [m*XSB_STRIDE + kb + 16]);
        avh[p] = cat8(*(const v8bf*)&Xdhi[m*XSB_STRIDE + kb],
                      *(const v8bf*)&Xdhi[m*XSB_STRIDE + kb + 16]);
        avl[p] = cat8(*(const v8bf*)&Xdlo[m*XSB_STRIDE + kb],
                      *(const v8bf*)&Xdlo[m*XSB_STRIDE + kb + 16]);
    }

    for (int t = 0; t < 4; ++t) {
        const int h0 = (wave * 4 + t) * 16;
        v8f accU = {0.f,0.f,0.f,0.f,0.f,0.f,0.f,0.f};
        v8f accV = {0.f,0.f,0.f,0.f,0.f,0.f,0.f,0.f};
        #pragma unroll
        for (int p = 0; p < DX / 32; ++p) {     // 2 K-panels of 32
            const int bbase = ((p * HH + h0 + m) << 5) + boff;
            v16bf bh = cat8(*(const v8bf*)&W1hi[bbase], *(const v8bf*)&W1hi[bbase + 16]);
            v16bf bl = cat8(*(const v8bf*)&W1lo[bbase], *(const v8bf*)&W1lo[bbase + 16]);
            // bf16x3: hi*hi + hi*lo + lo*hi
            accU = wmma_bf16_k32(axh[p], bh, accU);
            accU = wmma_bf16_k32(axh[p], bl, accU);
            accU = wmma_bf16_k32(axl[p], bh, accU);
            accV = wmma_bf16_k32(avh[p], bh, accV);
            accV = wmma_bf16_k32(avh[p], bl, accV);
            accV = wmma_bf16_k32(avl[p], bh, accV);
        }
        const float bias = b1[h0 + m];
        const int rbase = hiHalf ? 8 : 0;       // f32 C/D layout: VGPR i -> rows i, i+8
        #pragma unroll
        for (int i = 0; i < 8; ++i) {
            float u  = accU[i] + bias;
            float th = fast_tanh(u);
            float g  = (1.0f - th * th) * accV[i];
            __bf16 h, l; splitf(g, h, l);
            Ghi[(rbase + i) * GSB_STRIDE + h0 + m] = h;
            Glo[(rbase + i) * GSB_STRIDE + h0 + m] = l;
        }
    }
    __syncthreads();

    // ---------------- Stage B: D = G * W2 (bf16x3 over 16 K-panels) ----------------
    const int n0 = wave * 16;
    v8f accD = {0.f,0.f,0.f,0.f,0.f,0.f,0.f,0.f};
    #pragma unroll 4
    for (int p = 0; p < HH / 32; ++p) {
        const int kb = p * 32 + aoff;
        v16bf agh = cat8(*(const v8bf*)&Ghi[m*GSB_STRIDE + kb],
                         *(const v8bf*)&Ghi[m*GSB_STRIDE + kb + 16]);
        v16bf agl = cat8(*(const v8bf*)&Glo[m*GSB_STRIDE + kb],
                         *(const v8bf*)&Glo[m*GSB_STRIDE + kb + 16]);
        const int bbase = ((p * DZ + n0 + m) << 5) + boff;
        v16bf bh = cat8(*(const v8bf*)&W2hi[bbase], *(const v8bf*)&W2hi[bbase + 16]);
        v16bf bl = cat8(*(const v8bf*)&W2lo[bbase], *(const v8bf*)&W2lo[bbase + 16]);
        accD = wmma_bf16_k32(agh, bh, accD);
        accD = wmma_bf16_k32(agh, bl, accD);
        accD = wmma_bf16_k32(agl, bh, accD);
    }

    // ---------------- Stage C: S = z*A^T + y*B^T in exact f32 WMMA ----------------
    const int koff = hiHalf ? 2 : 0;
    v8f accS = {0.f,0.f,0.f,0.f,0.f,0.f,0.f,0.f};
    #pragma unroll 8
    for (int k0 = 0; k0 < DZ; k0 += 4) {
        const int k = k0 + koff;
        v2f a = *(const v2f*)&Zs[m * ZS_STRIDE + k];
        v2f bw;
        bw.x = Amat[(n0 + m) * DZ + k + 0];     // B'[k][n] = A[n][k]
        bw.y = Amat[(n0 + m) * DZ + k + 1];
        accS = wmma_f32_k4(a, bw, accS);
    }
    #pragma unroll
    for (int k0 = 0; k0 < DY; k0 += 4) {
        const int k = k0 + koff;
        v2f a = *(const v2f*)&Ys[m * YS_STRIDE + k];
        v2f bw;
        bw.x = Bmat[(n0 + m) * DY + k + 0];     // B'[k][n] = B[n][k]
        bw.y = Bmat[(n0 + m) * DY + k + 1];
        accS = wmma_f32_k4(a, bw, accS);
    }

    // Per-row sum of squares of P = D - S over this wave's 16 columns.
    const int rbase = hiHalf ? 8 : 0;
    #pragma unroll
    for (int i = 0; i < 8; ++i) {
        float p  = accD[i] - accS[i];
        float sq = p * p;
        #pragma unroll
        for (int s = 1; s < 16; s <<= 1) sq += __shfl_xor(sq, s, 32);
        if ((lane & 15) == 0) atomicAdd(&rowsum[rbase + i], sq);
    }
    __syncthreads();

    if (tid < 32) {
        float r = (lane < 16) ? sqrtf(rowsum[lane]) : 0.0f;
        #pragma unroll
        for (int s = 1; s < 32; s <<= 1) r += __shfl_xor(r, s, 32);
        if (lane == 0) atomicAdd(out, r * (1.0f / (float)NN));
    }
}

// =======================================================================
// Pure-f32 WMMA fallback, used if d_ws is too small
// =======================================================================
#define XS_STRIDE 68
#define GS_STRIDE 520
__global__ __launch_bounds__(256) void pde_loss_fused_f32(
    const float* __restrict__ z_pred, const float* __restrict__ x_label,
    const float* __restrict__ y_label, const float* __restrict__ W1,
    const float* __restrict__ b1, const float* __restrict__ W2,
    const float* __restrict__ Amat, const float* __restrict__ Bmat,
    float* __restrict__ out)
{
    __shared__ float Xs [M_TILE * XS_STRIDE];
    __shared__ float Xds[M_TILE * XS_STRIDE];
    __shared__ float Gs [M_TILE * GS_STRIDE];
    __shared__ float Zs [M_TILE * ZS_STRIDE];
    __shared__ float Ys [M_TILE * YS_STRIDE];
    __shared__ float rowsum[M_TILE];

    const int tid  = threadIdx.x;
    const int wave = tid >> 5;
    const int lane = tid & 31;
    const int m    = lane & 15;
    const int koff = (lane < 16) ? 0 : 2;
    const int row0 = blockIdx.x * M_TILE;

    #pragma unroll
    for (int i = 0; i < 4; ++i) {
        int idx = tid + i * 256;
        int r = idx >> 6, c = idx & 63;
        float x = x_label[(row0 + r) * DX + c];
        Xs [r * XS_STRIDE + c] = x;
        Xds[r * XS_STRIDE + c] = fast_tanh(x);
    }
    #pragma unroll
    for (int i = 0; i < 8; ++i) {
        int idx = tid + i * 256;
        int r = idx >> 7, c = idx & 127;
        Zs[r * ZS_STRIDE + c] = z_pred[(row0 + r) * DZ + c];
    }
    { int r = tid >> 4, c = tid & 15;
      Ys[r * YS_STRIDE + c] = y_label[(row0 + r) * DY + c]; }
    if (tid < M_TILE) rowsum[tid] = 0.0f;
    __syncthreads();

    for (int t = 0; t < 4; ++t) {
        const int h0 = (wave * 4 + t) * 16;
        v8f accU = {0.f,0.f,0.f,0.f,0.f,0.f,0.f,0.f};
        v8f accV = {0.f,0.f,0.f,0.f,0.f,0.f,0.f,0.f};
        #pragma unroll
        for (int k0 = 0; k0 < DX; k0 += 4) {
            const int k = k0 + koff;
            v2f au = *(const v2f*)&Xs [m * XS_STRIDE + k];
            v2f av = *(const v2f*)&Xds[m * XS_STRIDE + k];
            v2f bw;
            bw.x = W1[(k + 0) * HH + h0 + m];
            bw.y = W1[(k + 1) * HH + h0 + m];
            accU = wmma_f32_k4(au, bw, accU);
            accV = wmma_f32_k4(av, bw, accV);
        }
        const float bias = b1[h0 + m];
        const int rbase = (lane < 16) ? 0 : 8;
        #pragma unroll
        for (int i = 0; i < 8; ++i) {
            float u  = accU[i] + bias;
            float th = fast_tanh(u);
            Gs[(rbase + i) * GS_STRIDE + h0 + m] = (1.0f - th * th) * accV[i];
        }
    }
    __syncthreads();

    const int n0 = wave * 16;
    v8f accD = {0.f,0.f,0.f,0.f,0.f,0.f,0.f,0.f};
    #pragma unroll 8
    for (int k0 = 0; k0 < HH; k0 += 4) {
        const int k = k0 + koff;
        v2f a = *(const v2f*)&Gs[m * GS_STRIDE + k];
        v2f bw;
        bw.x = W2[(k + 0) * DZ + n0 + m];
        bw.y = W2[(k + 1) * DZ + n0 + m];
        accD = wmma_f32_k4(a, bw, accD);
    }

    v8f accS = {0.f,0.f,0.f,0.f,0.f,0.f,0.f,0.f};
    #pragma unroll 8
    for (int k0 = 0; k0 < DZ; k0 += 4) {
        const int k = k0 + koff;
        v2f a = *(const v2f*)&Zs[m * ZS_STRIDE + k];
        v2f bw;
        bw.x = Amat[(n0 + m) * DZ + k + 0];
        bw.y = Amat[(n0 + m) * DZ + k + 1];
        accS = wmma_f32_k4(a, bw, accS);
    }
    #pragma unroll
    for (int k0 = 0; k0 < DY; k0 += 4) {
        const int k = k0 + koff;
        v2f a = *(const v2f*)&Ys[m * YS_STRIDE + k];
        v2f bw;
        bw.x = Bmat[(n0 + m) * DY + k + 0];
        bw.y = Bmat[(n0 + m) * DY + k + 1];
        accS = wmma_f32_k4(a, bw, accS);
    }

    const int rbase = (lane < 16) ? 0 : 8;
    #pragma unroll
    for (int i = 0; i < 8; ++i) {
        float p  = accD[i] - accS[i];
        float sq = p * p;
        #pragma unroll
        for (int s = 1; s < 16; s <<= 1) sq += __shfl_xor(sq, s, 32);
        if ((lane & 15) == 0) atomicAdd(&rowsum[rbase + i], sq);
    }
    __syncthreads();

    if (tid < 32) {
        float r = (lane < 16) ? sqrtf(rowsum[lane]) : 0.0f;
        #pragma unroll
        for (int s = 1; s < 32; s <<= 1) r += __shfl_xor(r, s, 32);
        if (lane == 0) atomicAdd(out, r * (1.0f / (float)NN));
    }
}

__global__ void zero_scalar(float* out) { out[0] = 0.0f; }

extern "C" void kernel_launch(void* const* d_in, const int* in_sizes, int n_in,
                              void* d_out, int out_size, void* d_ws, size_t ws_size,
                              hipStream_t stream) {
    (void)in_sizes; (void)n_in; (void)out_size;
    const float* z_pred  = (const float*)d_in[0];
    const float* x_label = (const float*)d_in[1];
    const float* y_label = (const float*)d_in[2];
    const float* W1      = (const float*)d_in[3];
    const float* b1      = (const float*)d_in[4];
    const float* W2      = (const float*)d_in[5];
    // d_in[6] = b2 (cancels in the JVP)
    const float* Amat    = (const float*)d_in[7];
    const float* Bmat    = (const float*)d_in[8];
    float* out = (float*)d_out;

    zero_scalar<<<1, 1, 0, stream>>>(out);

    // Workspace layout (bf16 elements): W1hi | W1lo | W2hi | W2lo
    const size_t w1e = (size_t)DX * HH;     // 32768
    const size_t w2e = (size_t)HH * DZ;     // 65536
    const size_t need = (2 * w1e + 2 * w2e) * sizeof(__bf16);  // 384 KiB

    if (ws_size >= need) {
        __bf16* W1hi = (__bf16*)d_ws;
        __bf16* W1lo = W1hi + w1e;
        __bf16* W2hi = W1lo + w1e;
        __bf16* W2lo = W2hi + w2e;
        pack_panels<<<(int)((w1e + 255) / 256), 256, 0, stream>>>(W1, W1hi, W1lo, DX, HH);
        pack_panels<<<(int)((w2e + 255) / 256), 256, 0, stream>>>(W2, W2hi, W2lo, HH, DZ);
        pde_loss_fused_bf16x3<<<NN / M_TILE, 256, 0, stream>>>(
            z_pred, x_label, y_label, b1, W1hi, W1lo, W2hi, W2lo, Amat, Bmat, out);
    } else {
        pde_loss_fused_f32<<<NN / M_TILE, 256, 0, stream>>>(
            z_pred, x_label, y_label, W1, b1, W2, Amat, Bmat, out);
    }
}
